// ImprovedInvertibleNet_5823975653647
// MI455X (gfx1250) — matrix operand order repaired and checked by reading, not compiled, not run on hardware
//
#include <hip/hip_runtime.h>
#include <math.h>

// ---------------------------------------------------------------------------
// ImprovedInvertibleNet forward for MI455X (gfx1250, wave32, WMMA).
//
// One wave32 owns 16 batch rows. The 16x128 activation tile lives in
// registers as four 16x32 f16 WMMA A-fragments. Each 128x128 GEMM is
// 8 N-tiles x 4 K-steps of v_wmma_f32_16x16x32_f16 (f32 accumulate).
// Epilogues run in the C layout (lane%16 = N column, VGPR v = M row) and
// round-trip through a per-wave LDS tile (row stride padded to 136 halves so
// the A-fragment gather is bank-conflict free). Weights are pre-packed once
// per launch into per-lane B-fragment order (f16) in the workspace.
// All transcendental math is branch-free (v_exp_f32 / v_rcp_f32) so the
// epilogue cannot stall the matrix pipe with EXEC-mask branching.
// ---------------------------------------------------------------------------

typedef __attribute__((ext_vector_type(16))) _Float16      v16h;
typedef __attribute__((ext_vector_type(8)))  float         v8f;
typedef __attribute__((ext_vector_type(8)))  unsigned int  v8u;

#define LAYERS          12
#define RSTEPS          3
#define HDIM            128
#define ALPHA_C         2.5f
#define EPS_C           1e-6f
#define LSTRIDE         136          // halves per LDS row (128 + 8 pad)
#define WAVES_PER_BLOCK 4
#define BLOCK_THREADS   (WAVES_PER_BLOCK * 32)
#define ROWS_PER_WAVE   16
#define FRAG_HALFS      512          // one 32x16 f16 B fragment
#define LOG2E_C         1.4426950408889634f

// branch-free tanh: tanh(z) = 1 - 2/(exp(2z)+1); exp via single v_exp_f32
__device__ __forceinline__ float tanh_fast(float z) {
    float e = __builtin_amdgcn_exp2f(z * (2.0f * LOG2E_C));
    return 1.0f - 2.0f * __builtin_amdgcn_rcpf(e + 1.0f);
}

__device__ __forceinline__ float exp_fast(float x) {
    return __builtin_amdgcn_exp2f(x * LOG2E_C);
}

__device__ __forceinline__ float gelu_f(float x) {
    float z = 0.7978845608028654f * (x + 0.044715f * x * x * x);
    return 0.5f * x * (1.0f + tanh_fast(z));
}

__device__ __forceinline__ void lds_fence() {
    // per-wave LDS producer->consumer ordering (DS ops are in-order per wave;
    // this makes the dependency explicit)
    asm volatile("s_wait_dscnt 0x0" ::: "memory");
}

__device__ __forceinline__ unsigned int pack2h(float a, float b) {
    union { _Float16 h[2]; unsigned int u; } t;
    t.h[0] = (_Float16)a;
    t.h[1] = (_Float16)b;
    return t.u;
}

// Gather one 16x32 f16 A-fragment per K-chunk from a row-major LDS tile.
// A layout used here (16-bit, 16x32, contiguous K): lanes 0-15 -> M, lane half selects the
// K = {0..15} vs {16..31} split (k = 16*hh + s), as in the packed B; VGPR j packs K pairs.
__device__ __forceinline__ void load_a_frags(const _Float16* buf, int mrow, int hh,
                                             v16h a[4]) {
#pragma unroll
    for (int kc = 0; kc < 4; ++kc) {
        union { v8u u; v16h h; } t;
#pragma unroll
        for (int j = 0; j < 8; ++j) {
            int klo = hh * 16 + 2 * j;
            t.u[j] = *(const unsigned int*)&buf[mrow * LSTRIDE + kc * 32 + klo];
        }
        a[kc] = t.h;
    }
}

// ---------------------------------------------------------------------------
// Pack Wa/Wb (f32 [L][R][128][128]) into f16 per-lane B-fragment order:
//   frag(l,r,mat,kc,nc): element (k = kc*32 + 16*(lane/16) + j, n = nc*16 + lane%16)
//   stored at frag_id*512 + lane*16 + j   (one aligned 32B load per lane).
// ---------------------------------------------------------------------------
__global__ void pack_weights_kernel(const float* __restrict__ Wa,
                                    const float* __restrict__ Wb,
                                    _Float16* __restrict__ dst, int total) {
    int idx = blockIdx.x * blockDim.x + threadIdx.x;
    if (idx >= total) return;
    int j    = idx & 15;  int t = idx >> 4;
    int lane = t & 31;    t >>= 5;
    int nc   = t & 7;     t >>= 3;
    int kc   = t & 3;     t >>= 2;
    int mat  = t & 1;     t >>= 1;
    int r    = t % 3;
    int l    = t / 3;
    int k = kc * 32 + (lane >> 4) * 16 + j;
    int n = nc * 16 + (lane & 15);
    size_t src = ((size_t)(l * 3 + r) * HDIM + k) * HDIM + n;
    dst[idx] = (_Float16)(mat ? Wb[src] : Wa[src]);
}

// ---------------------------------------------------------------------------
// Main fused flow kernel.
// ---------------------------------------------------------------------------
__launch_bounds__(BLOCK_THREADS)
__global__ void flow_kernel(const float* __restrict__ eta,
                            const float* __restrict__ an_log_s,
                            const float* __restrict__ an_b,
                            const float* __restrict__ W0,
                            const float* __restrict__ b0,
                            const float* __restrict__ ba,
                            const float* __restrict__ bb,
                            const float* __restrict__ Wout,
                            const float* __restrict__ bout,
                            const _Float16* __restrict__ wfrag,
                            float* __restrict__ out, int nrows) {
    __shared__ _Float16 smem[WAVES_PER_BLOCK * 2 * 16 * LSTRIDE];

    const int tid  = threadIdx.x;
    const int lane = tid & 31;
    const int wid  = tid >> 5;
    const int hh   = lane >> 4;     // half-wave select
    const int mrow = lane & 15;     // A-row / C-column index

    _Float16* hbuf = smem + wid * (2 * 16 * LSTRIDE);
    _Float16* ubuf = hbuf + 16 * LSTRIDE;

    const long r0 = ((long)blockIdx.x * WAVES_PER_BLOCK + wid) * ROWS_PER_WAVE;
    const long rr = r0 + mrow;
    const long rrc = (rr < nrows) ? rr : (long)nrows - 1;

    // ---- input transform: x = eta/ns * log1p(ns) ----
    float e0 = eta[2 * rrc + 0];
    float e1 = eta[2 * rrc + 1];
    float nrm = sqrtf(e0 * e0 + e1 * e1);
    float ns  = fmaxf(nrm, EPS_C);
    float sc  = log1pf(ns) / ns;
    float x1 = e0 * sc, x2 = e1 * sc;
    float ld = 0.0f;

    v16h a[4];

#pragma unroll 1
    for (int l = 0; l < LAYERS; ++l) {
        // ---- actnorm ----
        float als0 = an_log_s[2 * l + 0], als1 = an_log_s[2 * l + 1];
        float ab0  = an_b[2 * l + 0],     ab1  = an_b[2 * l + 1];
        x1 = exp_fast(als0) * x1 + ab0;
        x2 = exp_fast(als1) * x2 + ab1;
        ld += als0 + als1;

        // ---- h0 = gelu(x1 * W0 + b0): build row-major f16 tile in LDS ----
        {
            const float4* w4 = (const float4*)(W0 + (size_t)l * HDIM);
            const float4* b4 = (const float4*)(b0 + (size_t)l * HDIM);
#pragma unroll 4
            for (int cc = 0; cc < 16; ++cc) {
                int c4 = hh * 16 + cc;           // group of 4 columns
                float4 w = w4[c4];
                float4 b = b4[c4];
                float g0 = gelu_f(x1 * w.x + b.x);
                float g1 = gelu_f(x1 * w.y + b.y);
                float g2 = gelu_f(x1 * w.z + b.z);
                float g3 = gelu_f(x1 * w.w + b.w);
                uint2 p;
                p.x = pack2h(g0, g1);
                p.y = pack2h(g2, g3);
                *(uint2*)&hbuf[mrow * LSTRIDE + c4 * 4] = p;
            }
        }
        lds_fence();
        load_a_frags(hbuf, mrow, hh, a);

        // ---- 3 residual steps, each = two 128x128 WMMA GEMMs ----
#pragma unroll 1
        for (int r = 0; r < RSTEPS; ++r) {
            const _Float16* wa = wfrag + (size_t)(((l * 3 + r) * 2 + 0) * 32) * FRAG_HALFS;
            const _Float16* wb = wfrag + (size_t)(((l * 3 + r) * 2 + 1) * 32) * FRAG_HALFS;
            __builtin_prefetch(wb, 0, 0);                    // global_prefetch_b8
            const float* bap = ba + (l * 3 + r) * HDIM;
            const float* bbp = bb + (l * 3 + r) * HDIM;

            // GEMM1: u = gelu(h @ Wa + ba)  -> ubuf
#pragma unroll
            for (int nc = 0; nc < 8; ++nc) {
                v8f c = {0.f, 0.f, 0.f, 0.f, 0.f, 0.f, 0.f, 0.f};
#pragma unroll
                for (int kc = 0; kc < 4; ++kc) {
                    v16h bfr = *(const v16h*)(wa + (size_t)(kc * 8 + nc) * FRAG_HALFS + lane * 16);
                    c = __builtin_amdgcn_wmma_f32_16x16x32_f16(
                            false, a[kc], false, bfr, (short)0, c, false, false);
                }
                float biasv = bap[nc * 16 + mrow];
#pragma unroll
                for (int v = 0; v < 8; ++v) {
                    int m = v + 8 * hh;
                    ubuf[m * LSTRIDE + nc * 16 + mrow] = (_Float16)gelu_f(c[v] + biasv);
                }
            }
            lds_fence();
            load_a_frags(ubuf, mrow, hh, a);

            // GEMM2: hb = u @ Wb + bb ; h = gelu(h + hb) -> hbuf
#pragma unroll
            for (int nc = 0; nc < 8; ++nc) {
                v8f c = {0.f, 0.f, 0.f, 0.f, 0.f, 0.f, 0.f, 0.f};
#pragma unroll
                for (int kc = 0; kc < 4; ++kc) {
                    v16h bfr = *(const v16h*)(wb + (size_t)(kc * 8 + nc) * FRAG_HALFS + lane * 16);
                    c = __builtin_amdgcn_wmma_f32_16x16x32_f16(
                            false, a[kc], false, bfr, (short)0, c, false, false);
                }
                float biasv = bbp[nc * 16 + mrow];
#pragma unroll
                for (int v = 0; v < 8; ++v) {
                    int m = v + 8 * hh;
                    float hold = (float)hbuf[m * LSTRIDE + nc * 16 + mrow];
                    hbuf[m * LSTRIDE + nc * 16 + mrow] =
                        (_Float16)gelu_f(hold + c[v] + biasv);
                }
            }
            lds_fence();
            if (r < RSTEPS - 1) load_a_frags(hbuf, mrow, hh, a);
        }

        // ---- st = h @ Wout + bout (128 -> 2); lane half = output column ----
        float acc = 0.0f;
        {
            const float2* wo2 = (const float2*)(Wout + (size_t)l * HDIM * 2);
#pragma unroll
            for (int k0 = 0; k0 < HDIM; k0 += 8) {
                union { uint4 u; _Float16 h[8]; } t;
                t.u = *(const uint4*)&hbuf[mrow * LSTRIDE + k0];
#pragma unroll
                for (int j = 0; j < 8; ++j) {
                    float2 w = wo2[k0 + j];
                    acc += (float)t.h[j] * (hh ? w.y : w.x);
                }
            }
            acc += bout[2 * l + hh];
        }
        float st0 = __shfl(acc, mrow, 32);        // o = 0 from lane mrow
        float st1 = __shfl(acc, mrow + 16, 32);   // o = 1 from lane mrow+16

        // ---- coupling + logdet + swap ----
        float log_s = tanh_fast(st0 * (1.0f / ALPHA_C)) * ALPHA_C;
        float x2n = x2 * exp_fast(log_s) + st1;
        ld += log_s;
        float nx1 = x2n;     // y = [x1, x2n][:, ::-1]
        x2 = x1;
        x1 = nx1;
    }

    if (lane < 16 && rr < nrows) {
        out[2 * rr + 0] = x1;
        out[2 * rr + 1] = x2;
        out[2 * (long)nrows + rr] = ld;
    }
}

// ---------------------------------------------------------------------------
extern "C" void kernel_launch(void* const* d_in, const int* in_sizes, int n_in,
                              void* d_out, int out_size, void* d_ws, size_t ws_size,
                              hipStream_t stream) {
    const float* eta      = (const float*)d_in[0];
    const float* an_log_s = (const float*)d_in[1];
    const float* an_b     = (const float*)d_in[2];
    const float* W0       = (const float*)d_in[3];
    const float* b0       = (const float*)d_in[4];
    const float* Wa       = (const float*)d_in[5];
    const float* ba       = (const float*)d_in[6];
    const float* Wb       = (const float*)d_in[7];
    const float* bb       = (const float*)d_in[8];
    const float* Wout     = (const float*)d_in[9];
    const float* bout     = (const float*)d_in[10];
    float* out            = (float*)d_out;
    _Float16* wfrag       = (_Float16*)d_ws;   // 12*3*2*4*8*512 halves = 2.25 MB

    int nrows = in_sizes[0] / 2;

    int total = LAYERS * RSTEPS * 2 * 4 * 8 * 32 * 16;   // 1,179,648 fragment halves
    pack_weights_kernel<<<(total + 255) / 256, 256, 0, stream>>>(Wa, Wb, wfrag, total);

    int nwaves  = (nrows + ROWS_PER_WAVE - 1) / ROWS_PER_WAVE;
    int nblocks = (nwaves + WAVES_PER_BLOCK - 1) / WAVES_PER_BLOCK;
    flow_kernel<<<nblocks, BLOCK_THREADS, 0, stream>>>(
        eta, an_log_s, an_b, W0, b0, ba, bb, Wout, bout, wfrag, out, nrows);
}
